// PointNet2SemSegSSG_35966056136720
// MI455X (gfx1250) — compile-verified, matrix operand order checked by reading, not run on hardware
//
#include <hip/hip_runtime.h>

typedef __attribute__((ext_vector_type(16))) _Float16 v16h;
typedef __attribute__((ext_vector_type(8)))  float    v8f;
typedef _Float16 h16;

// ---------------------------------------------------------------------------
// Fragment-layout index helper.
// Activations X[c][n] (C padded to Cpad, multiple of 32) are stored as WMMA
// B-fragments: element index = ((ntile*KT + ktile)*32 + lane)*16 + e, with
//   ntile = n/16, lane = (n%16) + 16*((c%32)/8 % 2), e = (c%8) + 8*(c%32/16)
// so each lane's 16 halves for one (ntile,ktile) are 32 contiguous bytes.
// Weights use the identical layout with M in place of N (A-fragment).
// ---------------------------------------------------------------------------
__device__ __forceinline__ size_t frag_idx(int c, int n, int Cpad) {
  int ntile = n >> 4, nl = n & 15;
  int kt = c >> 5, kin = c & 31;
  int khalf = (kin >> 3) & 1;
  int e = (kin & 7) + ((kin >> 4) << 3);
  int lane = nl + (khalf << 4);
  return ((((size_t)ntile * (Cpad >> 5) + kt) << 5) + lane) * 16 + e;
}

// ---------------------------------------------------------------------------
// 0) Split pointcloud -> xyz (f32, B*N*3) and feats level0 (f16, [B][6][N])
// ---------------------------------------------------------------------------
__global__ void k_prep(const float* __restrict__ pc, float* __restrict__ xyz,
                       h16* __restrict__ f0, int B, int N) {
  int gid = blockIdx.x * blockDim.x + threadIdx.x;
  if (gid >= B * N) return;
  int b = gid / N, n = gid % N;
  const float* p = pc + (size_t)gid * 9;
  float* q = xyz + (size_t)gid * 3;
  q[0] = p[0]; q[1] = p[1]; q[2] = p[2];
  size_t base = (size_t)b * 6 * N + n;
#pragma unroll
  for (int c = 0; c < 6; ++c) f0[base + (size_t)c * N] = (h16)p[3 + c];
}

// ---------------------------------------------------------------------------
// 0b) Pack f32 weights (O x C) into f16 A-fragment layout (O x Cpad), zero
//     padded K. One thread per packed element; runs once per launch.
// ---------------------------------------------------------------------------
__global__ void k_pack_w(const float* __restrict__ W, h16* __restrict__ Wp,
                         int O, int C, int Cpad) {
  int gid = blockIdx.x * blockDim.x + threadIdx.x;
  if (gid >= O * Cpad) return;
  int e = gid & 15;
  int lane = (gid >> 4) & 31;
  int tile = gid >> 9;
  int KT = Cpad >> 5;
  int kt = tile % KT, mtile = tile / KT;
  int row = lane & 15, khalf = lane >> 4;
  int m = mtile * 16 + row;
  int k = kt * 32 + (e & 7) + 8 * khalf + 16 * (e >> 3);
  Wp[gid] = (k < C) ? (h16)W[(size_t)m * C + k] : (h16)0.0f;
}

// ---------------------------------------------------------------------------
// 1) Farthest point sampling. One block per batch; dists live in LDS.
// ---------------------------------------------------------------------------
__global__ void __launch_bounds__(256)
k_fps(const float* __restrict__ xyz, float* __restrict__ new_xyz, int N, int S) {
  __shared__ float sd[8192];      // min-squared-distance per point (32 KB)
  __shared__ float rv[256];
  __shared__ int   ri[256];
  __shared__ int   s_cur;
  int b = blockIdx.x;
  const float* P = xyz + (size_t)b * N * 3;
  float* Q = new_xyz + (size_t)b * S * 3;
  int t = threadIdx.x, nt = blockDim.x;
  for (int i = t; i < N; i += nt) sd[i] = 1e10f;
  if (t == 0) s_cur = 0;
  __syncthreads();
  for (int it = 0; it < S; ++it) {
    int cur = s_cur;
    float cx = P[cur * 3 + 0], cy = P[cur * 3 + 1], cz = P[cur * 3 + 2];
    if (t == 0) { Q[it * 3 + 0] = cx; Q[it * 3 + 1] = cy; Q[it * 3 + 2] = cz; }
    float bv = -1.0f; int bi = 0;
    for (int i = t; i < N; i += nt) {
      float dx = P[i * 3 + 0] - cx, dy = P[i * 3 + 1] - cy, dz = P[i * 3 + 2] - cz;
      float d = dx * dx + dy * dy + dz * dz;
      float old = sd[i];
      float nd = d < old ? d : old;
      sd[i] = nd;
      if (nd > bv) { bv = nd; bi = i; }     // strict > keeps first index
    }
    rv[t] = bv; ri[t] = bi;
    __syncthreads();
    for (int off = nt >> 1; off > 0; off >>= 1) {
      if (t < off) {
        float v2 = rv[t + off]; int i2 = ri[t + off];
        if (v2 > rv[t] || (v2 == rv[t] && i2 < ri[t])) { rv[t] = v2; ri[t] = i2; }
      }
      __syncthreads();
    }
    if (t == 0) s_cur = ri[0];
    __syncthreads();
  }
}

// ---------------------------------------------------------------------------
// 2) Ball query + grouping -> fragment-layout X [B][Cpad][S*32]:
//    rows 0..2 relative xyz, rows 3..3+Cf gathered features, rest zero.
// ---------------------------------------------------------------------------
__global__ void k_ball_group(const float* __restrict__ xyz, const float* __restrict__ nxyz,
                             const h16* __restrict__ feats, h16* __restrict__ X,
                             int B, int N, int S, int Cf, int Cpad, float r2) {
  int gid = blockIdx.x * blockDim.x + threadIdx.x;
  if (gid >= B * S) return;
  int b = gid / S, s = gid % S;
  const float* P = xyz + (size_t)b * N * 3;
  const float* q = nxyz + ((size_t)b * S + s) * 3;
  float qx = q[0], qy = q[1], qz = q[2];
  int idx[32]; int cnt = 0;
  for (int i = 0; i < N; ++i) {                    // index order == sorted order
    float dx = P[i * 3 + 0] - qx, dy = P[i * 3 + 1] - qy, dz = P[i * 3 + 2] - qz;
    if (dx * dx + dy * dy + dz * dz <= r2) { idx[cnt++] = i; if (cnt == 32) break; }
  }
  int fill = cnt ? idx[0] : 0;
  for (int j = cnt; j < 32; ++j) idx[j] = fill;
  int NC = S * 32;
  h16* Xb = X + (size_t)b * Cpad * NC;
  int n0 = s * 32;
#pragma unroll
  for (int j = 0; j < 32; ++j) {
    int i = idx[j];
    Xb[frag_idx(0, n0 + j, Cpad)] = (h16)(P[i * 3 + 0] - qx);
    Xb[frag_idx(1, n0 + j, Cpad)] = (h16)(P[i * 3 + 1] - qy);
    Xb[frag_idx(2, n0 + j, Cpad)] = (h16)(P[i * 3 + 2] - qz);
  }
  const h16* F = feats + (size_t)b * Cf * N;
  for (int c = 0; c < Cf; ++c)
    for (int j = 0; j < 32; ++j)
      Xb[frag_idx(3 + c, n0 + j, Cpad)] = F[(size_t)c * N + idx[j]];
  for (int c = Cf + 3; c < Cpad; ++c)             // zero K padding
    for (int j = 0; j < 32; ++j)
      Xb[frag_idx(c, n0 + j, Cpad)] = (h16)0.0f;
}

// ---------------------------------------------------------------------------
// 3) Fused GEMM + ReLU on v_wmma_f32_16x16x32_f16, fragment-native I/O.
//    One wave computes a 16(M) x 64(N) strip. Software-pipelined: the A + 4 B
//    fragments of K-step kt+1 are prefetched before the 4 WMMAs of step kt,
//    so WMMA latency overlaps the next loads instead of draining loadcnt.
//    mode 0: write f16 fragment layout (CpadOut = O, O multiple of 32)
//    mode 1: write f16 plain [O][Ncols]    mode 2: write f32 plain [O][Ncols]
// ---------------------------------------------------------------------------
__global__ void __launch_bounds__(32)
k_wmma_gemm_relu(const h16* __restrict__ Wp, const h16* __restrict__ X,
                 h16* __restrict__ Yh, float* __restrict__ Yf,
                 int O, int Cpad, int Ncols, int mode) {
  const int lane = threadIdx.x;
  const int KT = Cpad >> 5;
  const int mtile = blockIdx.y;
  const int nt0 = blockIdx.x * 4;
  const v16h* Aall = (const v16h*)Wp;
  const v16h* Ball = (const v16h*)(X + (size_t)blockIdx.z * Cpad * Ncols);
  size_t ai = ((size_t)mtile * KT) * 32 + lane;
  size_t b0 = ((size_t)(nt0 + 0) * KT) * 32 + lane;
  size_t b1 = ((size_t)(nt0 + 1) * KT) * 32 + lane;
  size_t b2 = ((size_t)(nt0 + 2) * KT) * 32 + lane;
  size_t b3 = ((size_t)(nt0 + 3) * KT) * 32 + lane;
  v8f acc0 = {}, acc1 = {}, acc2 = {}, acc3 = {};
  // prologue: load K-step 0
  v16h a  = Aall[ai];
  v16h x0 = Ball[b0];
  v16h x1 = Ball[b1];
  v16h x2 = Ball[b2];
  v16h x3 = Ball[b3];
  for (int kt = 0; kt < KT - 1; ++kt) {
    size_t o = (size_t)(kt + 1) * 32;
    // prefetch next K-step while current WMMAs execute
    v16h an  = Aall[ai + o];
    v16h y0  = Ball[b0 + o];
    v16h y1  = Ball[b1 + o];
    v16h y2  = Ball[b2 + o];
    v16h y3  = Ball[b3 + o];
    acc0 = __builtin_amdgcn_wmma_f32_16x16x32_f16(false, a, false, x0, (short)0, acc0, false, false);
    acc1 = __builtin_amdgcn_wmma_f32_16x16x32_f16(false, a, false, x1, (short)0, acc1, false, false);
    acc2 = __builtin_amdgcn_wmma_f32_16x16x32_f16(false, a, false, x2, (short)0, acc2, false, false);
    acc3 = __builtin_amdgcn_wmma_f32_16x16x32_f16(false, a, false, x3, (short)0, acc3, false, false);
    a = an; x0 = y0; x1 = y1; x2 = y2; x3 = y3;
  }
  acc0 = __builtin_amdgcn_wmma_f32_16x16x32_f16(false, a, false, x0, (short)0, acc0, false, false);
  acc1 = __builtin_amdgcn_wmma_f32_16x16x32_f16(false, a, false, x1, (short)0, acc1, false, false);
  acc2 = __builtin_amdgcn_wmma_f32_16x16x32_f16(false, a, false, x2, (short)0, acc2, false, false);
  acc3 = __builtin_amdgcn_wmma_f32_16x16x32_f16(false, a, false, x3, (short)0, acc3, false, false);

  const int row = lane & 15, khalf = lane >> 4;
  const size_t yb = (size_t)blockIdx.z * O * Ncols;
#pragma unroll
  for (int t = 0; t < 4; ++t) {
    const v8f& acc = (t == 0) ? acc0 : (t == 1) ? acc1 : (t == 2) ? acc2 : acc3;
    int n = (nt0 + t) * 16 + row;
#pragma unroll
    for (int r = 0; r < 8; ++r) {
      int o = mtile * 16 + r + 8 * khalf;
      float v = acc[r];
      v = v > 0.0f ? v : 0.0f;                 // fused ReLU
      if (mode == 0)      Yh[yb + frag_idx(o, n, O)] = (h16)v;
      else if (mode == 1) Yh[yb + (size_t)o * Ncols + n] = (h16)v;
      else                Yf[yb + (size_t)o * Ncols + n] = v;
    }
  }
}

// ---------------------------------------------------------------------------
// 4) Max over 32 grouped samples: fragment [B][O][S*32] -> plain [B][O][S]
// ---------------------------------------------------------------------------
__global__ void k_maxpool32(const h16* __restrict__ Y, h16* __restrict__ out,
                            int B, int O, int S) {
  int gid = blockIdx.x * blockDim.x + threadIdx.x;
  if (gid >= B * O * S) return;
  int s = gid % S; int o = (gid / S) % O; int b = gid / (S * O);
  const h16* Yb = Y + (size_t)b * O * ((size_t)S * 32);
  float mx = -3.4e38f;
#pragma unroll
  for (int j = 0; j < 32; ++j) {
    float v = (float)Yb[frag_idx(o, s * 32 + j, O)];
    mx = v > mx ? v : mx;
  }
  out[((size_t)b * O + o) * S + s] = (h16)mx;
}

// ---------------------------------------------------------------------------
// 5) FP: 3-NN inverse-distance interpolation + concat skip, writing fragment
//    layout Xcat [B][Cpad][Nu] (interp channels first, then skip, then zeros).
// ---------------------------------------------------------------------------
__global__ void k_interp_concat(const float* __restrict__ uxyz, const float* __restrict__ kxyz,
                                const h16* __restrict__ kfeat, const h16* __restrict__ skip,
                                h16* __restrict__ X, int B, int Nu, int Nk,
                                int Ck, int Cs, int Cpad) {
  int gid = blockIdx.x * blockDim.x + threadIdx.x;
  if (gid >= B * Nu) return;
  int b = gid / Nu, n = gid % Nu;
  const float* u = uxyz + ((size_t)b * Nu + n) * 3;
  const float* K = kxyz + (size_t)b * Nk * 3;
  float ux = u[0], uy = u[1], uz = u[2];
  float d0 = 3.4e38f, d1 = 3.4e38f, d2 = 3.4e38f;
  int i0 = 0, i1 = 0, i2 = 0;
  for (int i = 0; i < Nk; ++i) {
    float dx = K[i * 3 + 0] - ux, dy = K[i * 3 + 1] - uy, dz = K[i * 3 + 2] - uz;
    float d = dx * dx + dy * dy + dz * dz;
    if (d < d0)      { d2 = d1; i2 = i1; d1 = d0; i1 = i0; d0 = d; i0 = i; }
    else if (d < d1) { d2 = d1; i2 = i1; d1 = d;  i1 = i; }
    else if (d < d2) { d2 = d;  i2 = i; }
  }
  float w0 = 1.0f / (d0 + 1e-8f), w1 = 1.0f / (d1 + 1e-8f), w2 = 1.0f / (d2 + 1e-8f);
  float ws = w0 + w1 + w2; w0 /= ws; w1 /= ws; w2 /= ws;
  const h16* F = kfeat + (size_t)b * Ck * Nk;
  h16* Xb = X + (size_t)b * Cpad * Nu;
  for (int c = 0; c < Ck; ++c) {
    float v = w0 * (float)F[(size_t)c * Nk + i0]
            + w1 * (float)F[(size_t)c * Nk + i1]
            + w2 * (float)F[(size_t)c * Nk + i2];
    Xb[frag_idx(c, n, Cpad)] = (h16)v;
  }
  const h16* Sf = skip + (size_t)b * Cs * Nu;
  for (int c = 0; c < Cs; ++c)
    Xb[frag_idx(Ck + c, n, Cpad)] = Sf[(size_t)c * Nu + n];
  for (int c = Ck + Cs; c < Cpad; ++c)            // zero K padding
    Xb[frag_idx(c, n, Cpad)] = (h16)0.0f;
}

// ---------------------------------------------------------------------------
// Host orchestration
// ---------------------------------------------------------------------------
extern "C" void kernel_launch(void* const* d_in, const int* in_sizes, int n_in,
                              void* d_out, int out_size, void* d_ws, size_t ws_size,
                              hipStream_t stream) {
  (void)in_sizes; (void)n_in; (void)out_size; (void)ws_size;
  const int B = 8;
  const int N0 = 8192, N1 = 1024, N2 = 256, N3 = 64, N4 = 16;
  auto pad32 = [](int c) { return (c + 31) & ~31; };
  auto cdiv  = [](int a, int b) { return (a + b - 1) / b; };

  // ---- workspace carve-out (256B aligned) ----
  char* w = (char*)d_ws;
  size_t off = 0;
  auto alloc = [&](size_t bytes) -> void* {
    void* p = w + off;
    off = (off + bytes + 255) & ~(size_t)255;
    return p;
  };
  float* xyz0 = (float*)alloc((size_t)B * N0 * 3 * 4);
  float* xyz1 = (float*)alloc((size_t)B * N1 * 3 * 4);
  float* xyz2 = (float*)alloc((size_t)B * N2 * 3 * 4);
  float* xyz3 = (float*)alloc((size_t)B * N3 * 3 * 4);
  float* xyz4 = (float*)alloc((size_t)B * N4 * 3 * 4);
  h16* f0 = (h16*)alloc((size_t)B * 6   * N0 * 2);
  h16* f1 = (h16*)alloc((size_t)B * 64  * N1 * 2);
  h16* f2 = (h16*)alloc((size_t)B * 128 * N2 * 2);
  h16* f3 = (h16*)alloc((size_t)B * 256 * N3 * 2);
  h16* f4 = (h16*)alloc((size_t)B * 512 * N4 * 2);
  h16* g3 = (h16*)alloc((size_t)B * 256 * N3 * 2);   // FP4 out
  h16* g2 = (h16*)alloc((size_t)B * 256 * N2 * 2);   // FP3 out
  h16* g1 = (h16*)alloc((size_t)B * 128 * N1 * 2);   // FP2 out
  const size_t PP = (size_t)B * 64 * 32768;          // worst-stage frag elems
  h16* bufA = (h16*)alloc(PP * 2);
  h16* bufB = (h16*)alloc(PP * 2);

  // ---- pack all 21 weight matrices into A-fragment f16 layout ----
  struct WSpec { int O, C; };
  const WSpec ws13[13] = {
    {32, 9}, {32, 32}, {64, 32},          // sa1  (d_in 1..3)
    {64, 67}, {64, 64}, {128, 64},        // sa2  (d_in 4..6)
    {128, 131}, {128, 128}, {256, 128},   // sa3  (d_in 7..9)
    {256, 259}, {256, 256}, {512, 256},   // sa4  (d_in 10..12)
    {128, 134}                            // fp1 layer1 (d_in 13)
  };
  h16* wp[22];
  for (int i = 0; i < 13; ++i) {
    int O = ws13[i].O, C = ws13[i].C, Cp = pad32(C);
    wp[i + 1] = (h16*)alloc((size_t)O * Cp * 2);
    k_pack_w<<<cdiv(O * Cp, 256), 256, 0, stream>>>((const float*)d_in[i + 1], wp[i + 1], O, C, Cp);
  }
  const WSpec ws9[8] = {
    {128, 128}, {128, 128},               // fp1 layers 2,3 (d_in 14,15)
    {256, 320}, {128, 256},               // fp2 (d_in 16,17)
    {256, 384}, {256, 256},               // fp3 (d_in 18,19)
    {256, 768}, {256, 256}                // fp4 (d_in 20,21)
  };
  for (int i = 0; i < 8; ++i) {
    int O = ws9[i].O, C = ws9[i].C, Cp = pad32(C);
    wp[i + 14] = (h16*)alloc((size_t)O * Cp * 2);
    k_pack_w<<<cdiv(O * Cp, 256), 256, 0, stream>>>((const float*)d_in[i + 14], wp[i + 14], O, C, Cp);
  }

  auto gemm = [&](const h16* Wp, const h16* Xp, void* Y, int O, int Cpad, int Ncols, int mode) {
    dim3 g(Ncols / 64, O / 16, B);
    k_wmma_gemm_relu<<<g, 32, 0, stream>>>(Wp, Xp, (h16*)Y, (float*)Y, O, Cpad, Ncols, mode);
  };

  // ---- prep ----
  k_prep<<<cdiv(B * N0, 256), 256, 0, stream>>>((const float*)d_in[0], xyz0, f0, B, N0);

  // ---- SA stages ----
  struct SA { float* xi; int Ni; float* xo; int S; h16* fi; int Cf;
              float r2; int wbase; int o1, o2, o3; h16* fo; };
  const SA sa[4] = {
    { xyz0, N0, xyz1, N1, f0, 6,   0.01f, 1,  32,  32,  64,  f1 },
    { xyz1, N1, xyz2, N2, f1, 64,  0.04f, 4,  64,  64,  128, f2 },
    { xyz2, N2, xyz3, N3, f2, 128, 0.16f, 7,  128, 128, 256, f3 },
    { xyz3, N3, xyz4, N4, f3, 256, 0.64f, 10, 256, 256, 512, f4 },
  };
  for (int l = 0; l < 4; ++l) {
    const SA& m = sa[l];
    int NC = m.S * 32;
    int Cp0 = pad32(m.Cf + 3);
    k_fps<<<B, 256, 0, stream>>>(m.xi, m.xo, m.Ni, m.S);
    k_ball_group<<<cdiv(B * m.S, 256), 256, 0, stream>>>(
        m.xi, m.xo, m.fi, bufA, B, m.Ni, m.S, m.Cf, Cp0, m.r2);
    gemm(wp[m.wbase + 0], bufA, bufB, m.o1, Cp0,  NC, 0);
    gemm(wp[m.wbase + 1], bufB, bufA, m.o2, m.o1, NC, 0);
    gemm(wp[m.wbase + 2], bufA, bufB, m.o3, m.o2, NC, 0);
    k_maxpool32<<<cdiv(B * m.o3 * m.S, 256), 256, 0, stream>>>(bufB, m.fo, B, m.o3, m.S);
  }

  // ---- FP4: (xyz3 <- xyz4), kfeat f4 (512), skip f3 (256) -> g3 plain ----
  k_interp_concat<<<cdiv(B * N3, 256), 256, 0, stream>>>(
      xyz3, xyz4, f4, f3, bufA, B, N3, N4, 512, 256, 768);
  gemm(wp[20], bufA, bufB, 256, 768, N3, 0);
  gemm(wp[21], bufB, g3,   256, 256, N3, 1);

  // ---- FP3: (xyz2 <- xyz3), kfeat g3 (256), skip f2 (128) -> g2 plain ----
  k_interp_concat<<<cdiv(B * N2, 256), 256, 0, stream>>>(
      xyz2, xyz3, g3, f2, bufA, B, N2, N3, 256, 128, 384);
  gemm(wp[18], bufA, bufB, 256, 384, N2, 0);
  gemm(wp[19], bufB, g2,   256, 256, N2, 1);

  // ---- FP2: (xyz1 <- xyz2), kfeat g2 (256), skip f1 (64) -> g1 plain ----
  k_interp_concat<<<cdiv(B * N1, 256), 256, 0, stream>>>(
      xyz1, xyz2, g2, f1, bufA, B, N1, N2, 256, 64, 320);
  gemm(wp[16], bufA, bufB, 256, 320, N1, 0);
  gemm(wp[17], bufB, g1,   128, 256, N1, 1);

  // ---- FP1: (xyz0 <- xyz1), kfeat g1 (128), skip f0 (6) -> d_out f32 ----
  k_interp_concat<<<cdiv(B * N0, 256), 256, 0, stream>>>(
      xyz0, xyz1, g1, f0, bufA, B, N0, N1, 128, 6, 160);
  gemm(wp[13], bufA, bufB, 128, 160, N0, 0);
  gemm(wp[14], bufB, bufA, 128, 128, N0, 0);
  gemm(wp[15], bufA, d_out, 128, 128, N0, 2);
}
